// PokeTransformer_54915451846956
// MI455X (gfx1250) — matrix operand order repaired and checked by reading, not compiled, not run on hardware
//
#include <hip/hip_runtime.h>
#include <hip/hip_bf16.h>
#include <stdint.h>

// ---- problem constants (match reference) ----
#define L_    4
#define D_    512
#define H_    8
#define HD    64
#define FF_   1024
#define B_    8
#define S_    480
#define TPT   15
#define CAPV  6.0f
#define NEGV  (-1e30f)

typedef __attribute__((ext_vector_type(16))) _Float16 v16h;
typedef __attribute__((ext_vector_type(8)))  _Float16 v8h;
typedef __attribute__((ext_vector_type(8)))  float    v8f;

// ------------------------------------------------------------------
// WMMA fragment helpers (wave32, 16x16x32 f16 -> f32)
// A layout (16-bit A 16x32): lane<16 holds row=lane, K={0..7,16..23};
//                            lane>=16 holds row=lane-16, K={8..15,24..31}
// B layout (16-bit B 32x16): lane<16 holds col=lane,   K={0..7,16..23};
//                            lane>=16 holds col=lane-16, K={8..15,24..31}
// C/D (f32 16x16): vgpr v: lanes0-15 -> (M=v, N=lane); lanes16-31 -> (M=8+v, N=lane-16)
// ------------------------------------------------------------------
static __device__ __forceinline__ v16h load_a_frag(const _Float16* A, int lda,
                                                   int row0, int k0, int lane) {
  const int half = lane >> 4;
  const int r    = lane & 15;
  const _Float16* p = A + (size_t)(row0 + r) * lda + k0 + half * 8;
  v8h lo = *(const v8h*)(p);
  v8h hi = *(const v8h*)(p + 16);
  v16h a;
#pragma unroll
  for (int i = 0; i < 8; ++i) { a[i] = lo[i]; a[8 + i] = hi[i]; }
  return a;
}

// BT is the B matrix stored transposed: BT[n][k], leading dim ldbt (K contiguous)
static __device__ __forceinline__ v16h load_bt_frag(const _Float16* BT, int ldbt,
                                                    int n0, int k0, int lane) {
  const int kh = lane >> 4;
  const int c  = lane & 15;
  const _Float16* p = BT + (size_t)(n0 + c) * ldbt + k0 + kh * 8;
  v8h lo = *(const v8h*)(p);
  v8h hi = *(const v8h*)(p + 16);
  v16h b;
#pragma unroll
  for (int i = 0; i < 8; ++i) { b[i] = lo[i]; b[8 + i] = hi[i]; }
  return b;
}

static __device__ __forceinline__ v8f wmma16(v16h a, v16h b, v8f c) {
  return __builtin_amdgcn_wmma_f32_16x16x32_f16(false, a, false, b, (short)0, c,
                                                false, false);
}

static __device__ __forceinline__ void zero8(v8f& x) {
#pragma unroll
  for (int i = 0; i < 8; ++i) x[i] = 0.0f;
}

// ------------------------------------------------------------------
// Weight cast + transpose: W [Ln][K][N] f32  ->  WT [Ln][N][K] f16
// ------------------------------------------------------------------
__global__ void wtrans_kernel(const float* __restrict__ W, _Float16* __restrict__ WT,
                              int Ln, int K, int N) {
  size_t idx = (size_t)blockIdx.x * blockDim.x + threadIdx.x;
  size_t total = (size_t)Ln * K * N;
  if (idx >= total) return;
  int n = (int)(idx % N);
  size_t t = idx / N;
  int kk = (int)(t % K);
  int l  = (int)(t / K);
  WT[((size_t)l * N + n) * K + kk] = (_Float16)W[idx];
}

__global__ void copy_f32_kernel(const float* __restrict__ src, float* __restrict__ dst, int n) {
  int i = blockIdx.x * blockDim.x + threadIdx.x;
  if (i < n) dst[i] = src[i];
}

// ------------------------------------------------------------------
// LayerNorm: X f32 [rows, D] -> out f16 [rows, D]
// ------------------------------------------------------------------
__global__ void ln_kernel(const float* __restrict__ X, const float* __restrict__ g,
                          const float* __restrict__ bt, _Float16* __restrict__ out) {
  __shared__ float s1[256];
  __shared__ float s2[256];
  const int row = blockIdx.x;
  const int t = threadIdx.x;
  const float* xr = X + (size_t)row * D_;
  float a = xr[t];
  float b = xr[t + 256];
  s1[t] = a + b;
  s2[t] = a * a + b * b;
  __syncthreads();
#pragma unroll
  for (int o = 128; o > 0; o >>= 1) {
    if (t < o) { s1[t] += s1[t + o]; s2[t] += s2[t + o]; }
    __syncthreads();
  }
  const float mean = s1[0] * (1.0f / D_);
  const float var  = s2[0] * (1.0f / D_) - mean * mean;
  const float inv  = rsqrtf(var + 1e-5f);
  out[(size_t)row * D_ + t]       = (_Float16)((a - mean) * inv * g[t] + bt[t]);
  out[(size_t)row * D_ + t + 256] = (_Float16)((b - mean) * inv * g[t + 256] + bt[t + 256]);
}

// ------------------------------------------------------------------
// Core 32(M) x 64(N) tile accumulation: 8 WMMAs per K-step, B frags
// shared across the two M-rows. acc[mi][t], mi in {0,1}, t in {0..3}.
// ------------------------------------------------------------------
static __device__ __forceinline__ void gemm_tile_32x64(const _Float16* __restrict__ A,
                                                       const _Float16* __restrict__ BT,
                                                       int row0, int n0, int K, int lane,
                                                       v8f acc[2][4]) {
  for (int k0 = 0; k0 < K; k0 += 32) {
    v16h a0 = load_a_frag(A, K, row0,      k0, lane);
    v16h a1 = load_a_frag(A, K, row0 + 16, k0, lane);
#pragma unroll
    for (int t = 0; t < 4; ++t) {
      v16h bf = load_bt_frag(BT, K, n0 + 16 * t, k0, lane);
      acc[0][t] = wmma16(a0, bf, acc[0][t]);
      acc[1][t] = wmma16(a1, bf, acc[1][t]);
    }
  }
}

// ------------------------------------------------------------------
// GEMM (QKV): C[M=B*S, N=3D] = A(h16) @ WqkvT^T + bias, scattered into
//   q [B,H,S,64] f16, k [B,H,S,64] f16, vT [B,H,64,S] f16
// grid: (N/64, M/32), block: 32 (1 wave)
// ------------------------------------------------------------------
__global__ void gemm_qkv_kernel(const _Float16* __restrict__ A, const _Float16* __restrict__ BT,
                                const float* __restrict__ bias,
                                _Float16* __restrict__ q, _Float16* __restrict__ k,
                                _Float16* __restrict__ vT) {
  const int lane = threadIdx.x;
  const int n0   = blockIdx.x * 64;
  const int row0 = blockIdx.y * 32;
  v8f acc[2][4];
#pragma unroll
  for (int mi = 0; mi < 2; ++mi)
#pragma unroll
    for (int t = 0; t < 4; ++t) zero8(acc[mi][t]);

  gemm_tile_32x64(A, BT, row0, n0, D_, lane, acc);

  const int c = lane & 15;
  const int rbase = (lane >> 4) * 8;
#pragma unroll
  for (int mi = 0; mi < 2; ++mi) {
#pragma unroll
    for (int t = 0; t < 4; ++t) {
#pragma unroll
      for (int vv = 0; vv < 8; ++vv) {
        const int m = row0 + mi * 16 + rbase + vv;
        const int n = n0 + 16 * t + c;
        const float val = acc[mi][t][vv] + bias[n];
        const int bb = m / S_, s = m % S_;
        const int which = n >> 9;          // 0:q 1:k 2:v
        const int j = n & 511;
        const int hh = j >> 6, d = j & 63;
        if (which == 0)
          q[(((size_t)bb * H_ + hh) * S_ + s) * HD + d] = (_Float16)val;
        else if (which == 1)
          k[(((size_t)bb * H_ + hh) * S_ + s) * HD + d] = (_Float16)val;
        else
          vT[(((size_t)bb * H_ + hh) * HD + d) * S_ + s] = (_Float16)val;
      }
    }
  }
}

// ------------------------------------------------------------------
// GEMM with residual add: X[m,n] += A @ BT^T + bias   (X f32, ld = N)
// grid: (N/64, M/32), block 32
// ------------------------------------------------------------------
__global__ void gemm_res_kernel(const _Float16* __restrict__ A, const _Float16* __restrict__ BT,
                                const float* __restrict__ bias, float* __restrict__ X,
                                int N, int K) {
  const int lane = threadIdx.x;
  const int n0   = blockIdx.x * 64;
  const int row0 = blockIdx.y * 32;
  v8f acc[2][4];
#pragma unroll
  for (int mi = 0; mi < 2; ++mi)
#pragma unroll
    for (int t = 0; t < 4; ++t) zero8(acc[mi][t]);

  gemm_tile_32x64(A, BT, row0, n0, K, lane, acc);

  const int c = lane & 15;
  const int rbase = (lane >> 4) * 8;
#pragma unroll
  for (int mi = 0; mi < 2; ++mi) {
#pragma unroll
    for (int t = 0; t < 4; ++t) {
#pragma unroll
      for (int vv = 0; vv < 8; ++vv) {
        const int m = row0 + mi * 16 + rbase + vv;
        const int n = n0 + 16 * t + c;
        X[(size_t)m * N + n] += acc[mi][t][vv] + bias[n];
      }
    }
  }
}

// ------------------------------------------------------------------
// GEMM with exact GELU epilogue -> f16 out [M, N]
// grid: (N/64, M/32), block 32
// ------------------------------------------------------------------
__global__ void gemm_gelu_kernel(const _Float16* __restrict__ A, const _Float16* __restrict__ BT,
                                 const float* __restrict__ bias, _Float16* __restrict__ O,
                                 int N, int K) {
  const int lane = threadIdx.x;
  const int n0   = blockIdx.x * 64;
  const int row0 = blockIdx.y * 32;
  v8f acc[2][4];
#pragma unroll
  for (int mi = 0; mi < 2; ++mi)
#pragma unroll
    for (int t = 0; t < 4; ++t) zero8(acc[mi][t]);

  gemm_tile_32x64(A, BT, row0, n0, K, lane, acc);

  const int c = lane & 15;
  const int rbase = (lane >> 4) * 8;
#pragma unroll
  for (int mi = 0; mi < 2; ++mi) {
#pragma unroll
    for (int t = 0; t < 4; ++t) {
#pragma unroll
      for (int vv = 0; vv < 8; ++vv) {
        const int m = row0 + mi * 16 + rbase + vv;
        const int n = n0 + 16 * t + c;
        float x = acc[mi][t][vv] + bias[n];
        x = 0.5f * x * (1.0f + erff(x * 0.70710678118654752f));
        O[(size_t)m * N + n] = (_Float16)x;
      }
    }
  }
}

// ------------------------------------------------------------------
// Fused attention: per (b, h, 16-row q tile), one wave.
//  scores (WMMA) -> scale -> tanh cap -> episodic mask -> softmax (LDS)
//  -> f16 probs -> attn @ V (WMMA) -> o16 [B*S, D]
// grid: (S/16, H, B), block 32
// ------------------------------------------------------------------
__global__ void attn_kernel(const _Float16* __restrict__ q, const _Float16* __restrict__ k,
                            const _Float16* __restrict__ vT,
                            const int* __restrict__ eids, const int* __restrict__ npf_p,
                            _Float16* __restrict__ o16) {
  __shared__ __align__(16) float    sS[16][S_];
  __shared__ __align__(16) _Float16 sP[16][S_];

  const int lane = threadIdx.x;
  const int qt = blockIdx.x, hh = blockIdx.y, bb = blockIdx.z;
  const int row0 = qt * 16;
  const int npf = npf_p[0];

  const _Float16* qb  = q  + (((size_t)bb * H_ + hh) * S_) * HD;
  const _Float16* kb  = k  + (((size_t)bb * H_ + hh) * S_) * HD;
  const _Float16* vTb = vT + (((size_t)bb * H_ + hh) * HD) * S_;
  const int* eb = eids + bb * S_;

  const int c = lane & 15;
  const int rbase = (lane >> 4) * 8;

  // ---- scores: q fragments are loop-invariant, hoist them ----
  const v16h qf0 = load_a_frag(qb, HD, row0, 0,  lane);
  const v16h qf1 = load_a_frag(qb, HD, row0, 32, lane);

  for (int kt = 0; kt < S_ / 16; ++kt) {
    v8f acc; zero8(acc);
    {
      v16h bf0 = load_bt_frag(kb, HD, kt * 16, 0,  lane);  // k rows act as B^T
      v16h bf1 = load_bt_frag(kb, HD, kt * 16, 32, lane);
      acc = wmma16(qf0, bf0, acc);
      acc = wmma16(qf1, bf1, acc);
    }
#pragma unroll
    for (int vv = 0; vv < 8; ++vv) {
      const int r  = rbase + vv;
      const int qi = row0 + r;
      const int ki = kt * 16 + c;
      float sv = acc[vv] * 0.125f;               // 1/sqrt(64)
      sv = CAPV * tanhf(sv * (1.0f / CAPV));     // logit soft cap
      const int qturn = qi / TPT, kturn = ki / TPT, kidx = ki % TPT;
      const bool same_turn  = (qturn == kturn);
      const bool prev_field = (kidx == 0) && (kturn < qturn) && (qturn - kturn <= npf);
      const bool same_ep    = (eb[qi] == eb[ki]);
      sS[r][ki] = (same_ep && (same_turn || prev_field)) ? sv : NEGV;
    }
  }
  __syncthreads();

  // ---- per-row softmax over S_ columns ----
  for (int r = 0; r < 16; ++r) {
    float mx = -3.0e38f;
    for (int cc = lane; cc < S_; cc += 32) mx = fmaxf(mx, sS[r][cc]);
#pragma unroll
    for (int o = 16; o >= 1; o >>= 1) mx = fmaxf(mx, __shfl_xor(mx, o, 32));
    float sum = 0.0f;
    for (int cc = lane; cc < S_; cc += 32) {
      float e = expf(sS[r][cc] - mx);
      sS[r][cc] = e;
      sum += e;
    }
#pragma unroll
    for (int o = 16; o >= 1; o >>= 1) sum += __shfl_xor(sum, o, 32);
    const float inv = 1.0f / sum;
    for (int cc = lane; cc < S_; cc += 32) sP[r][cc] = (_Float16)(sS[r][cc] * inv);
  }
  __syncthreads();

  // ---- attn @ V ----
  v8f oacc[4];
#pragma unroll
  for (int t = 0; t < 4; ++t) zero8(oacc[t]);
  const _Float16* Pp = (const _Float16*)&sP[0][0];
  for (int k0 = 0; k0 < S_; k0 += 32) {
    v16h af = load_a_frag(Pp, S_, 0, k0, lane);
#pragma unroll
    for (int t = 0; t < 4; ++t) {
      v16h bf = load_bt_frag(vTb, S_, 16 * t, k0, lane);  // vT[d][s]: K contiguous
      oacc[t] = wmma16(af, bf, oacc[t]);
    }
  }
#pragma unroll
  for (int t = 0; t < 4; ++t) {
#pragma unroll
    for (int vv = 0; vv < 8; ++vv) {
      const int r = rbase + vv;
      const int d = 16 * t + c;
      o16[(size_t)(bb * S_ + row0 + r) * D_ + hh * HD + d] = (_Float16)oacc[t][vv];
    }
  }
}

// ------------------------------------------------------------------
// host-side orchestration
// ------------------------------------------------------------------
extern "C" void kernel_launch(void* const* d_in, const int* in_sizes, int n_in,
                              void* d_out, int out_size, void* d_ws, size_t ws_size,
                              hipStream_t stream) {
  (void)in_sizes; (void)n_in; (void)out_size; (void)ws_size;

  const float* x_in = (const float*)d_in[0];
  const int*   eids = (const int*)  d_in[1];
  const int*   npf  = (const int*)  d_in[2];
  const float* Wqkv = (const float*)d_in[3];
  const float* bqkv = (const float*)d_in[4];
  const float* Wo   = (const float*)d_in[5];
  const float* bo   = (const float*)d_in[6];
  const float* g1   = (const float*)d_in[7];
  const float* b1   = (const float*)d_in[8];
  const float* g2   = (const float*)d_in[9];
  const float* b2   = (const float*)d_in[10];
  const float* W1   = (const float*)d_in[11];
  const float* bf1  = (const float*)d_in[12];
  const float* W2   = (const float*)d_in[13];
  const float* bf2  = (const float*)d_in[14];
  float* X = (float*)d_out;   // residual stream lives in d_out

  // ---- carve up workspace (f16 halves, 256B aligned) ----
  uintptr_t base = (uintptr_t)d_ws;
  size_t off = 0;
  auto take = [&](size_t halves) -> _Float16* {
    _Float16* p = (_Float16*)(base + off);
    off += (halves * sizeof(_Float16) + 255) & ~(size_t)255;
    return p;
  };
  _Float16* WqkvT = take((size_t)L_ * 3 * D_ * D_);
  _Float16* WoT   = take((size_t)L_ * D_ * D_);
  _Float16* W1T   = take((size_t)L_ * FF_ * D_);
  _Float16* W2T   = take((size_t)L_ * D_ * FF_);
  _Float16* h16   = take((size_t)B_ * S_ * D_);
  _Float16* q16   = take((size_t)B_ * H_ * S_ * HD);
  _Float16* k16   = take((size_t)B_ * H_ * S_ * HD);
  _Float16* vT16  = take((size_t)B_ * H_ * HD * S_);
  _Float16* o16   = take((size_t)B_ * S_ * D_);
  _Float16* t16   = take((size_t)B_ * S_ * FF_);

  const int M = B_ * S_;        // 3840

  // ---- weight cast/transpose (once per launch; deterministic) ----
  {
    size_t tot;
    tot = (size_t)L_ * D_ * 3 * D_;
    wtrans_kernel<<<(unsigned)((tot + 255) / 256), 256, 0, stream>>>(Wqkv, WqkvT, L_, D_, 3 * D_);
    tot = (size_t)L_ * D_ * D_;
    wtrans_kernel<<<(unsigned)((tot + 255) / 256), 256, 0, stream>>>(Wo, WoT, L_, D_, D_);
    tot = (size_t)L_ * D_ * FF_;
    wtrans_kernel<<<(unsigned)((tot + 255) / 256), 256, 0, stream>>>(W1, W1T, L_, D_, FF_);
    tot = (size_t)L_ * FF_ * D_;
    wtrans_kernel<<<(unsigned)((tot + 255) / 256), 256, 0, stream>>>(W2, W2T, L_, FF_, D_);
  }

  // ---- init residual stream ----
  {
    int n = M * D_;
    copy_f32_kernel<<<(n + 255) / 256, 256, 0, stream>>>(x_in, X, n);
  }

  // ---- layers ----
  for (int l = 0; l < L_; ++l) {
    const _Float16* wqkv_l = WqkvT + (size_t)l * 3 * D_ * D_;
    const _Float16* wo_l   = WoT   + (size_t)l * D_ * D_;
    const _Float16* w1_l   = W1T   + (size_t)l * FF_ * D_;
    const _Float16* w2_l   = W2T   + (size_t)l * D_ * FF_;

    // h = LN1(x) -> f16
    ln_kernel<<<M, 256, 0, stream>>>(X, g1 + l * D_, b1 + l * D_, h16);

    // qkv = h @ Wqkv + bqkv  (scatter to q/k/vT)
    gemm_qkv_kernel<<<dim3(3 * D_ / 64, M / 32), 32, 0, stream>>>(
        h16, wqkv_l, bqkv + l * 3 * D_, q16, k16, vT16);

    // fused attention -> o16
    attn_kernel<<<dim3(S_ / 16, H_, B_), 32, 0, stream>>>(q16, k16, vT16, eids, npf, o16);

    // x += o16 @ Wo + bo
    gemm_res_kernel<<<dim3(D_ / 64, M / 32), 32, 0, stream>>>(
        o16, wo_l, bo + l * D_, X, D_, D_);

    // h2 = LN2(x) -> f16
    ln_kernel<<<M, 256, 0, stream>>>(X, g2 + l * D_, b2 + l * D_, h16);

    // t = gelu(h2 @ W1 + bf1) -> f16
    gemm_gelu_kernel<<<dim3(FF_ / 64, M / 32), 32, 0, stream>>>(
        h16, w1_l, bf1 + l * FF_, t16, FF_, D_);

    // x += t @ W2 + bf2
    gemm_res_kernel<<<dim3(D_ / 64, M / 32), 32, 0, stream>>>(
        t16, w2_l, bf2 + l * D_, X, D_, FF_);
  }
}
